// BSGMP_36532991820475
// MI455X (gfx1250) — matrix operand order repaired and checked by reading, not compiled, not run on hardware
//
#include <hip/hip_runtime.h>

typedef __attribute__((ext_vector_type(16))) _Float16 v16h;
typedef __attribute__((ext_vector_type(8)))  _Float16 v8h;
typedef __attribute__((ext_vector_type(8)))  float    v8f;

#define LDIM 64
#define KP_EDGE 160   // 132 padded to multiple of 32
#define KP_NODE 128

// ---------------------------------------------------------------------------
// WMMA building blocks
// ---------------------------------------------------------------------------

// Stage W[K x 64] (f32, row-major) into LDS pre-swizzled fragment-major f16:
//   Ws[((kt*4 + t)*32 + lane)*16 + e] = w[(kt*32 + e + 16*(lane>>4)) * 64 + t*16 + (lane&15)]
// so each lane's B fragment is 32 contiguous bytes -> 2x ds_load_b128.
template<int KA>
__device__ __forceinline__ void stage_w_frag(_Float16* Ws, const float* __restrict__ w,
                                             int Kvalid)
{
    for (int idx = threadIdx.x; idx < KA * LDIM; idx += blockDim.x) {
        int e    = idx & 15;
        int lane = (idx >> 4) & 31;
        int tile = idx >> 9;
        int t    = tile & 3;
        int kt   = tile >> 2;
        int m16  = lane & 15, hl = lane >> 4;
        int k = kt * 32 + e + 16 * hl;   // B k order used here (same as A): K = e + 16*lanehalf
        int n = t * 16 + m16;            //           N = lane&15 (+ 16*tile)
        Ws[idx] = (k < Kvalid) ? (_Float16)w[k * LDIM + n] : (_Float16)0.0f;
    }
}

// One wave computes rows [rowBase, rowBase+16) x all 64 cols of As[rows x KA] * W.
// As is row-major f16; Ws is fragment-major (see stage_w_frag).
// relu_to_ys != 0: y = relu(y+bias) -> Ys (f16).  relu_to_ys == 0: y = y+bias -> yf (f32).
template<int KA>
__device__ __forceinline__ void gemm_stripe(const _Float16* As, const _Float16* Ws,
                                            const float* __restrict__ bias,
                                            _Float16* Ys, float* yf, int relu_to_ys,
                                            int rowBase, int m16, int hl, int lane)
{
    v8f acc[4] = {};
    const v16h* Wf = (const v16h*)Ws;
#pragma unroll
    for (int kt = 0; kt < KA / 32; ++kt) {
        // A k order used here (same as B): elements 0..15 -> K = k0 + 16*hl + 0..15
        const _Float16* ap = As + (rowBase + m16) * KA + kt * 32 + 16 * hl;
        v8h alo = *(const v8h*)(ap);        // 16B aligned: KA rows are 128/256/320B
        v8h ahi = *(const v8h*)(ap + 8);
        v16h a = __builtin_shufflevector(alo, ahi,
                                         0, 1, 2, 3, 4, 5, 6, 7,
                                         8, 9, 10, 11, 12, 13, 14, 15);
#pragma unroll
        for (int t = 0; t < 4; ++t) {
            v16h b = Wf[(kt * 4 + t) * 32 + lane];   // 32B contiguous per lane
            acc[t] = __builtin_amdgcn_wmma_f32_16x16x32_f16(
                false, a, false, b, (short)0, acc[t], false, false);
        }
    }
#pragma unroll
    for (int t = 0; t < 4; ++t) {
        int n = t * 16 + m16;
        float bb = bias[n];
#pragma unroll
        for (int r = 0; r < 8; ++r) {      // C/D layout: VGPR r -> M = r + 8*lanehalf
            int m = rowBase + r + 8 * hl;
            float v = acc[t][r] + bb;
            if (relu_to_ys) { v = fmaxf(v, 0.0f); Ys[m * LDIM + n] = (_Float16)v; }
            else            { yf[m * LDIM + n] = v; }
        }
    }
}

// 3-layer MLP + layernorm over a 64-row tile staged in As. Result left in yf (64x64 f32).
// Requires blockDim.x == 128 (4 waves; each owns a private 16-row stripe).
template<int KP>
__device__ __forceinline__ void mlp3_tile(
    _Float16* As, _Float16* Ws, _Float16* Ys, float* yf,
    const float* w0, const float* b0, const float* w1, const float* b1,
    const float* w2, const float* b2, const float* gamma, const float* beta, int K0)
{
    const int tid  = threadIdx.x;
    const int lane = tid & 31;
    const int m16  = lane & 15;
    const int hl   = lane >> 4;
    const int rowBase = (tid >> 5) * 16;

    stage_w_frag<KP>(Ws, w0, K0);
    __syncthreads();
    gemm_stripe<KP>(As, Ws, b0, Ys, yf, 1, rowBase, m16, hl, lane);
    __syncthreads();

    stage_w_frag<64>(Ws, w1, 64);
    __syncthreads();
    gemm_stripe<64>(Ys, Ws, b1, Ys, yf, 1, rowBase, m16, hl, lane);
    __syncthreads();

    stage_w_frag<64>(Ws, w2, 64);
    __syncthreads();
    gemm_stripe<64>(Ys, Ws, b2, Ys, yf, 0, rowBase, m16, hl, lane);  // raw f32 into yf
    __syncthreads();

    // layernorm: 2 threads per row, partner combine via shfl_xor(1); float4 LDS accesses
    int r = tid >> 1, hs = tid & 1;
    float4* row = (float4*)(yf + r * 64 + hs * 32);       // 16B aligned
    const float4* g4 = (const float4*)(gamma + hs * 32);
    const float4* b4 = (const float4*)(beta + hs * 32);
    float s = 0.f, s2 = 0.f;
#pragma unroll
    for (int q = 0; q < 8; ++q) {
        float4 v = row[q];
        s  += v.x + v.y + v.z + v.w;
        s2 += v.x * v.x + v.y * v.y + v.z * v.z + v.w * v.w;
    }
    s  += __shfl_xor(s, 1, 32);
    s2 += __shfl_xor(s2, 1, 32);
    float mu  = s  * (1.0f / 64.0f);
    float var = s2 * (1.0f / 64.0f) - mu * mu;
    float rs  = rsqrtf(var + 1e-5f);
#pragma unroll
    for (int q = 0; q < 8; ++q) {
        float4 v = row[q], gg = g4[q], bb = b4[q];
        v.x = (v.x - mu) * rs * gg.x + bb.x;
        v.y = (v.y - mu) * rs * gg.y + bb.y;
        v.z = (v.z - mu) * rs * gg.z + bb.z;
        v.w = (v.w - mu) * rs * gg.w + bb.w;
        row[q] = v;
    }
    __syncthreads();
}

// ---------------------------------------------------------------------------
// Fused GMP kernels
// ---------------------------------------------------------------------------

// Edge MLP: gather [dir(3), |dir|, x[i](64), x[j](64)] -> 3-layer MLP -> layernorm
// -> atomic scatter-add into aggr[j].  64 edges per block.
__global__ __launch_bounds__(128)
void k_edge_mlp(const float* __restrict__ h, const float* __restrict__ pos,
                const int* __restrict__ gi, const int* __restrict__ gj, int E,
                const float* w0, const float* b0, const float* w1, const float* b1,
                const float* w2, const float* b2, const float* gg, const float* bt,
                float* __restrict__ aggr)
{
    __shared__ __align__(32) _Float16 smem[64 * KP_EDGE + KP_EDGE * LDIM + 64 * LDIM];
    _Float16* As = smem;
    _Float16* Ws = smem + 64 * KP_EDGE;
    _Float16* Ys = Ws + KP_EDGE * LDIM;
    float* yf = (float*)smem;   // aliases As (20480B >= 16384B), dead after layer 0

    const int tid   = threadIdx.x;
    const int eBase = blockIdx.x * 64;

    __builtin_prefetch(w0, 0, 2);   // global_prefetch path

    for (int idx = tid; idx < 64 * KP_EDGE; idx += blockDim.x) {
        int r = idx / KP_EDGE, c = idx % KP_EDGE;
        int e = eBase + r;
        float v = 0.0f;
        if (e < E) {
            int i = gi[e], j = gj[e];
            if (c < 3) {
                v = pos[i * 3 + c] - pos[j * 3 + c];
            } else if (c == 3) {
                float dx = pos[i * 3 + 0] - pos[j * 3 + 0];
                float dy = pos[i * 3 + 1] - pos[j * 3 + 1];
                float dz = pos[i * 3 + 2] - pos[j * 3 + 2];
                v = sqrtf(dx * dx + dy * dy + dz * dz);
            } else if (c < 68) {
                v = h[i * 64 + (c - 4)];
            } else if (c < 132) {
                v = h[j * 64 + (c - 68)];
            }
        }
        As[idx] = (_Float16)v;
    }
    __syncthreads();

    mlp3_tile<KP_EDGE>(As, Ws, Ys, yf, w0, b0, w1, b1, w2, b2, gg, bt, 132);

    for (int idx = tid; idx < 64 * 64; idx += blockDim.x) {
        int r = idx >> 6, c = idx & 63;
        int e = eBase + r;
        if (e < E) atomicAdd(&aggr[gj[e] * 64 + c], yf[idx]);
    }
}

// Node MLP: [x(64) | aggr(64)] -> MLP -> layernorm -> +x residual -> out.
__global__ __launch_bounds__(128)
void k_node_mlp(const float* __restrict__ x, const float* __restrict__ aggr, int N,
                const float* w0, const float* b0, const float* w1, const float* b1,
                const float* w2, const float* b2, const float* gg, const float* bt,
                float* __restrict__ out)
{
    __shared__ __align__(32) _Float16 smem[64 * KP_NODE + KP_NODE * LDIM + 64 * LDIM];
    _Float16* As = smem;
    _Float16* Ws = smem + 64 * KP_NODE;
    _Float16* Ys = Ws + KP_NODE * LDIM;
    float* yf = (float*)smem;   // aliases As (16384B == 16384B)

    const int tid   = threadIdx.x;
    const int nBase = blockIdx.x * 64;

    __builtin_prefetch(w0, 0, 2);

    for (int idx = tid; idx < 64 * KP_NODE; idx += blockDim.x) {
        int r = idx >> 7, c = idx & 127;
        int n = nBase + r;
        float v = 0.0f;
        if (n < N) v = (c < 64) ? x[n * 64 + c] : aggr[n * 64 + (c - 64)];
        As[idx] = (_Float16)v;
    }
    __syncthreads();

    mlp3_tile<KP_NODE>(As, Ws, Ys, yf, w0, b0, w1, b1, w2, b2, gg, bt, 128);

    for (int idx = tid; idx < 64 * 64; idx += blockDim.x) {
        int r = idx >> 6, c = idx & 63;
        int n = nBase + r;
        if (n < N) out[n * 64 + c] = yf[idx] + x[n * 64 + c];
    }
}

// ---------------------------------------------------------------------------
// Scalar helper kernels
// ---------------------------------------------------------------------------

__global__ void k_zero(float* p, int n)   { int i = blockIdx.x*blockDim.x+threadIdx.x; if (i < n) p[i] = 0.0f; }
__global__ void k_fill1(float* p, int n)  { int i = blockIdx.x*blockDim.x+threadIdx.x; if (i < n) p[i] = 1.0f; }
__global__ void k_copy(float* d, const float* s, int n) { int i = blockIdx.x*blockDim.x+threadIdx.x; if (i < n) d[i] = s[i]; }
__global__ void k_add (float* d, const float* s, int n) { int i = blockIdx.x*blockDim.x+threadIdx.x; if (i < n) d[i] += s[i]; }
__global__ void k_addeps(float* d, int n) { int i = blockIdx.x*blockDim.x+threadIdx.x; if (i < n) d[i] += 1e-12f; }

__global__ void k_deg(const int* gi, int E, float* deg) {
    int e = blockIdx.x*blockDim.x+threadIdx.x;
    if (e < E) atomicAdd(&deg[gi[e]], 1.0f);
}
__global__ void k_wsend(const float* w, const int* gi, const int* gj, const float* deg,
                        int E, float* wsd, float* agw) {
    int e = blockIdx.x*blockDim.x+threadIdx.x;
    if (e < E) { float v = w[gi[e]] / deg[gi[e]]; wsd[e] = v; atomicAdd(&agw[gj[e]], v); }
}
__global__ void k_ec(const float* wsd, const int* gj, const float* agw, int E, float* ec) {
    int e = blockIdx.x*blockDim.x+threadIdx.x;
    if (e < E) ec[e] = wsd[e] / agw[gj[e]];
}
__global__ void k_edgeconv(const float* x, int C, const int* gi, const int* gj,
                           const float* ew, float* out, int E, int agg, int total) {
    int idx = blockIdx.x*blockDim.x+threadIdx.x;
    if (idx < total) {
        int e = idx / C, c = idx - e * C;
        int s = agg ? gi[e] : gj[e];
        int t = agg ? gj[e] : gi[e];
        atomicAdd(&out[t * C + c], x[s * C + c] * ew[e]);
    }
}
__global__ void k_gather(const float* in, const int* ids, int C, int total, float* out) {
    int idx = blockIdx.x*blockDim.x+threadIdx.x;
    if (idx < total) { int r = idx / C, c = idx - r * C; out[idx] = in[ids[r] * C + c]; }
}
__global__ void k_scatter_set(const float* in, const int* ids, int C, int total, float* out) {
    int idx = blockIdx.x*blockDim.x+threadIdx.x;
    if (idx < total) { int r = idx / C, c = idx - r * C; out[ids[r] * C + c] = in[idx]; }
}

// ---------------------------------------------------------------------------
// Host orchestration
// ---------------------------------------------------------------------------

extern "C" void kernel_launch(void* const* d_in, const int* in_sizes, int n_in,
                              void* d_out, int out_size, void* d_ws, size_t ws_size,
                              hipStream_t stream)
{
    (void)in_sizes; (void)n_in; (void)out_size; (void)ws_size;
    const int cN0 = 20000, cE0 = 160000;
    const int cN1 = 5000,  cE1 = 40000;
    const int cN2 = 1250,  cE2 = 10000;

    const float* h_in = (const float*)d_in[0];
    const float* pos0 = (const float*)d_in[1];
    const int* g0 = (const int*)d_in[2];
    const int* g1 = (const int*)d_in[3];
    const int* g2 = (const int*)d_in[4];
    const int* ids0 = (const int*)d_in[5];
    const int* ids1 = (const int*)d_in[6];
    const int *gi0 = g0, *gj0 = g0 + cE0;
    const int *gi1 = g1, *gj1 = g1 + cE1;
    const int *gi2 = g2, *gj2 = g2 + cE2;

    float* f = (float*)d_ws;
    auto carve = [&](long n) { float* p = f; f += n; return p; };
    float* hA   = carve((long)cN0 * 64);
    float* hB   = carve((long)cN0 * 64);
    float* aggr = carve((long)cN0 * 64);
    float* tmp  = carve((long)cN0 * 64);
    float* d0b  = carve((long)cN0 * 64);
    float* d1b  = carve((long)cN1 * 64);
    float* pos1 = carve((long)cN1 * 3);
    float* pos2 = carve((long)cN2 * 3);
    float* tpos = carve((long)cN0 * 3);
    float* wl   = carve(cN0);
    float* deg  = carve(cN0);
    float* agw  = carve(cN0);
    float* wsd  = carve(cE0);
    float* ec0  = carve(cE0);
    float* ec1  = carve(cE1);

    auto Z  = [&](float* p, long n) { k_zero<<<dim3((unsigned)((n + 255) / 256)), dim3(256), 0, stream>>>(p, (int)n); };
    auto CP = [&](float* d, const float* s, long n) { k_copy<<<dim3((unsigned)((n + 255) / 256)), dim3(256), 0, stream>>>(d, s, (int)n); };

    struct MLPP { const float *w0, *b0, *w1, *b1, *w2, *b2, *g, *bt; };
    auto mlpP = [&](int base) {
        MLPP p;  // sorted-key flatten order: b0,b1,b2,bt,g,w0,w1,w2
        p.b0 = (const float*)d_in[base + 0]; p.b1 = (const float*)d_in[base + 1];
        p.b2 = (const float*)d_in[base + 2]; p.bt = (const float*)d_in[base + 3];
        p.g  = (const float*)d_in[base + 4]; p.w0 = (const float*)d_in[base + 5];
        p.w1 = (const float*)d_in[base + 6]; p.w2 = (const float*)d_in[base + 7];
        return p;
    };

    float* cur = hA;
    float* alt = hB;
    CP(cur, h_in, (long)cN0 * 64);
    k_fill1<<<dim3((cN0 + 255) / 256), dim3(256), 0, stream>>>(wl, cN0);

    auto run_gmp = [&](int gbase, const float* posv, const int* gi, const int* gj, int N, int E) {
        MLPP pe = mlpP(gbase), pn = mlpP(gbase + 8);
        Z(aggr, (long)N * 64);
        k_edge_mlp<<<dim3((E + 63) / 64), dim3(128), 0, stream>>>(
            cur, posv, gi, gj, E,
            pe.w0, pe.b0, pe.w1, pe.b1, pe.w2, pe.b2, pe.g, pe.bt, aggr);
        k_node_mlp<<<dim3((N + 63) / 64), dim3(128), 0, stream>>>(
            cur, aggr, N,
            pn.w0, pn.b0, pn.w1, pn.b1, pn.w2, pn.b2, pn.g, pn.bt, alt);
        float* t = cur; cur = alt; alt = t;
    };
    auto run_stack = [&](int sbase, const float* posv, const int* gi, const int* gj, int N, int E) {
        run_gmp(sbase,      posv, gi, gj, N, E);
        run_gmp(sbase + 16, posv, gi, gj, N, E);
    };
    auto cal_ew = [&](const float* w, const int* gi, const int* gj, int N, int E, float* ec) {
        Z(deg, N); Z(agw, N);
        k_deg  <<<dim3((E + 255) / 256), dim3(256), 0, stream>>>(gi, E, deg);
        k_wsend<<<dim3((E + 255) / 256), dim3(256), 0, stream>>>(w, gi, gj, deg, E, wsd, agw);
        k_addeps<<<dim3((N + 255) / 256), dim3(256), 0, stream>>>(agw, N);
        k_ec   <<<dim3((E + 255) / 256), dim3(256), 0, stream>>>(wsd, gj, agw, E, ec);
    };
    auto econv = [&](const float* x, int C, const int* gi, const int* gj, const float* ew,
                     float* out, int N, int E, int agg) {
        Z(out, (long)N * C);
        int total = E * C;
        k_edgeconv<<<dim3((total + 255) / 256), dim3(256), 0, stream>>>(x, C, gi, gj, ew, out, E, agg, total);
    };

    // params flatten: 7 graph inputs, then sorted stacks: bottom, down0, down1, up0, up1
    const int PB = 7;
    const int S_bottom = PB, S_down0 = PB + 32, S_down1 = PB + 64, S_up0 = PB + 96, S_up1 = PB + 128;

    // ---- down level 0 ----
    run_stack(S_down0, pos0, gi0, gj0, cN0, cE0);
    CP(d0b, cur, (long)cN0 * 64);
    cal_ew(wl, gi0, gj0, cN0, cE0, ec0);
    econv(cur,  64, gi0, gj0, ec0, tmp,  cN0, cE0, 1);
    econv(pos0,  3, gi0, gj0, ec0, tpos, cN0, cE0, 1);
    k_gather<<<dim3((cN1 * 64 + 255) / 256), dim3(256), 0, stream>>>(tmp,  ids0, 64, cN1 * 64, alt);
    { float* t = cur; cur = alt; alt = t; }
    k_gather<<<dim3((cN1 * 3 + 255) / 256), dim3(256), 0, stream>>>(tpos, ids0, 3, cN1 * 3, pos1);
    k_gather<<<dim3((cN1 + 255) / 256),     dim3(256), 0, stream>>>(agw,  ids0, 1, cN1, wl);

    // ---- down level 1 ----
    run_stack(S_down1, pos1, gi1, gj1, cN1, cE1);
    CP(d1b, cur, (long)cN1 * 64);
    cal_ew(wl, gi1, gj1, cN1, cE1, ec1);
    econv(cur,  64, gi1, gj1, ec1, tmp,  cN1, cE1, 1);
    econv(pos1,  3, gi1, gj1, ec1, tpos, cN1, cE1, 1);
    k_gather<<<dim3((cN2 * 64 + 255) / 256), dim3(256), 0, stream>>>(tmp,  ids1, 64, cN2 * 64, alt);
    { float* t = cur; cur = alt; alt = t; }
    k_gather<<<dim3((cN2 * 3 + 255) / 256), dim3(256), 0, stream>>>(tpos, ids1, 3, cN2 * 3, pos2);

    // ---- bottom ----
    run_stack(S_bottom, pos2, gi2, gj2, cN2, cE2);

    // ---- up i=0 (level 1) ----
    Z(tmp, (long)cN1 * 64);
    k_scatter_set<<<dim3((cN2 * 64 + 255) / 256), dim3(256), 0, stream>>>(cur, ids1, 64, cN2 * 64, tmp);
    econv(tmp, 64, gi1, gj1, ec1, alt, cN1, cE1, 0);
    { float* t = cur; cur = alt; alt = t; }
    run_stack(S_up0, pos1, gi1, gj1, cN1, cE1);
    k_add<<<dim3((cN1 * 64 + 255) / 256), dim3(256), 0, stream>>>(cur, d1b, cN1 * 64);

    // ---- up i=1 (level 0) ----
    Z(tmp, (long)cN0 * 64);
    k_scatter_set<<<dim3((cN1 * 64 + 255) / 256), dim3(256), 0, stream>>>(cur, ids0, 64, cN1 * 64, tmp);
    econv(tmp, 64, gi0, gj0, ec0, alt, cN0, cE0, 0);
    { float* t = cur; cur = alt; alt = t; }
    run_stack(S_up1, pos0, gi0, gj0, cN0, cE0);
    k_add<<<dim3((cN0 * 64 + 255) / 256), dim3(256), 0, stream>>>(cur, d0b, cN0 * 64);

    CP((float*)d_out, cur, (long)cN0 * 64);
}
